// StringGNNResidualBilinearModel_6923487282435
// MI455X (gfx1250) — compile-verified, matrix operand order checked by reading, not run on hardware
//
#include <hip/hip_runtime.h>
#include <hip/hip_bf16.h>
#include <math.h>

// MI455X / gfx1250, wave32. All GEMMs via v_wmma_f32_16x16x32_f16 (f16 in, f32 acc).

typedef __attribute__((ext_vector_type(16))) _Float16 v16h;
typedef __attribute__((ext_vector_type(8)))  _Float16 v8h;
typedef __attribute__((ext_vector_type(8)))  float    v8f;

#define GNN   256
#define HDIM  512
#define MIDD  2048
#define NCLS  3
#define RNK   512
#define NGENE 6640
#define NLAYER 6
#define LDSW  40    // 32 + 8 halves pad

// K index inside a 16-half fragment, per ISA 05_wmma 16-bit A/B layout:
// elements 0..7 -> K = half*8 + e ; elements 8..15 -> K = 16 + half*8 + (e-8)
__device__ __forceinline__ int frag_k(int e, int hh) {
  int v = e >> 1;
  return ((v < 4) ? 0 : 16) + hh * 8 + (v & 3) * 2 + (e & 1);
}

__device__ __forceinline__ v8h cvt8(float4 x, float4 y) {
  v8h h;
  h[0] = (_Float16)x.x; h[1] = (_Float16)x.y; h[2] = (_Float16)x.z; h[3] = (_Float16)x.w;
  h[4] = (_Float16)y.x; h[5] = (_Float16)y.y; h[6] = (_Float16)y.z; h[7] = (_Float16)y.w;
  return h;
}

struct Staged { float4 ax, ay, wx, wy; };

// Generic Linear: C[M,N] = act(A[M,K](lda) * W[N,K]^T + bias[N] + res[M,N](ldres))
// act: 0=none, 1=relu, 2=exact-erf GELU.
// Block = 256 thr = 8 waves -> 64x64 tile; wave = 16x32 (1 A-frag x 2 B-frags, 2 WMMA).
// Requires: K % 32 == 0 (true for all shapes here), 16B-aligned rows (lda%4==0).
__global__ __launch_bounds__(256)
void linear_wmma(const float* __restrict__ A, int lda,
                 const float* __restrict__ W,
                 const float* __restrict__ bias,
                 const float* __restrict__ res, int ldres,
                 float* __restrict__ C, int ldc,
                 int M, int N, int K, int act)
{
  __shared__ __attribute__((aligned(16))) _Float16 Ah[2][64][LDSW];
  __shared__ __attribute__((aligned(16))) _Float16 Wh[2][64][LDSW];

  const int tid  = threadIdx.x;
  const int lane = tid & 31;
  const int wav  = tid >> 5;
  const int tm   = wav >> 1;          // 0..3 : 16-row subtile
  const int tn2  = wav & 1;           // 0..1 : pair of 16-col subtiles
  const int hh   = lane >> 4;         // lane half -> K split
  const int l15  = lane & 15;
  const int mBase = blockIdx.y * 64;
  const int nBase = blockIdx.x * 64;

  // staging: 64 rows x 32 K-cols per operand; 4 threads/row, 8 contiguous cols each
  const int  srow = tid >> 2;
  const int  scol = (tid & 3) * 8;
  const bool aOk  = (mBase + srow) < M;
  const bool wOk  = (nBase + srow) < N;
  const float* Arow = A + (size_t)min(mBase + srow, M - 1) * lda + scol;
  const float* Wrow = W + (size_t)min(nBase + srow, N - 1) * K   + scol;

  union { v8f v; float f[8]; } acc[2];
  #pragma unroll
  for (int r = 0; r < 8; ++r) { acc[0].f[r] = 0.f; acc[1].f[r] = 0.f; }

  auto load_tile = [&](int kk) -> Staged {
    Staged t;
    t.ax = make_float4(0.f,0.f,0.f,0.f); t.ay = t.ax; t.wx = t.ax; t.wy = t.ax;
    if (aOk) {
      t.ax = *(const float4*)(Arow + kk);
      t.ay = *(const float4*)(Arow + kk + 4);
    }
    if (wOk) {
      t.wx = *(const float4*)(Wrow + kk);
      t.wy = *(const float4*)(Wrow + kk + 4);
    }
    return t;
  };
  auto store_tile = [&](const Staged& t, int buf) {
    *(v8h*)&Ah[buf][srow][scol] = cvt8(t.ax, t.ay);
    *(v8h*)&Wh[buf][srow][scol] = cvt8(t.wx, t.wy);
  };

  {
    Staged t0s = load_tile(0);
    store_tile(t0s, 0);
  }
  __syncthreads();

  const int nsteps = K >> 5;
  for (int s = 0; s < nsteps; ++s) {
    const int buf  = s & 1;
    const bool more = (s + 1) < nsteps;

    // 1) issue next tile's global loads (consumed only after the WMMAs below,
    //    so s_wait_loadcnt can sink past the matrix ops)
    Staged nxt;
    if (more) nxt = load_tile((s + 1) << 5);

    // 2) fragments for this tile + 2 WMMAs
    union { v16h v; _Float16 f[16]; } af, bf0, bf1;
    #pragma unroll
    for (int e = 0; e < 16; ++e) {
      int k = frag_k(e, hh);
      af.f[e]  = Ah[buf][tm * 16 + l15][k];            // A row = M-lane
      bf0.f[e] = Wh[buf][(tn2 * 2 + 0) * 16 + l15][k]; // B col = N-lane (B[k][n]=W[n][k])
      bf1.f[e] = Wh[buf][(tn2 * 2 + 1) * 16 + l15][k];
    }
    acc[0].v = __builtin_amdgcn_wmma_f32_16x16x32_f16(
        false, af.v, false, bf0.v, (short)0, acc[0].v, false, false);
    acc[1].v = __builtin_amdgcn_wmma_f32_16x16x32_f16(
        false, af.v, false, bf1.v, (short)0, acc[1].v, false, false);

    // 3) convert + stage next tile into the other LDS buffer
    if (more) store_tile(nxt, buf ^ 1);
    __syncthreads();
  }

  // D layout: lane half selects M+8; VGPR r = M row 0..7; lane&15 = N col
  #pragma unroll
  for (int j = 0; j < 2; ++j) {
    const int ncol = nBase + (tn2 * 2 + j) * 16 + l15;
    #pragma unroll
    for (int r = 0; r < 8; ++r) {
      const int mrow = mBase + tm * 16 + r + 8 * hh;
      if (mrow < M && ncol < N) {
        float v = acc[j].f[r];
        if (bias) v += bias[ncol];
        if (res)  v += res[(size_t)mrow * ldres + ncol];
        if (act == 1)      v = fmaxf(v, 0.f);
        else if (act == 2) v = 0.5f * v * (1.f + erff(v * 0.70710678118654752f));
        C[(size_t)mrow * ldc + ncol] = v;
      }
    }
  }
}

__global__ void zero_f32(float* __restrict__ p, long n) {
  long i = (long)blockIdx.x * blockDim.x + threadIdx.x;
  if (i < n) p[i] = 0.f;
}

// agg[dst] += cache[src] * w   (one thread per edge-channel)
__global__ void scatter_add(const int* __restrict__ src, const int* __restrict__ dst,
                            const float* __restrict__ ew, const float* __restrict__ cache,
                            float* __restrict__ agg, long total)
{
  long i = (long)blockIdx.x * blockDim.x + threadIdx.x;
  if (i >= total) return;
  int e = (int)(i >> 8);
  int d = (int)(i & 255);
  atomicAdd(&agg[(size_t)dst[e] * GNN + d], cache[(size_t)src[e] * GNN + d] * ew[e]);
}

// gather with fallback + LayerNorm(256). One wave per row, 8 elems/lane.
__global__ __launch_bounds__(256)
void gather_ln256(const int* __restrict__ idx, const float* __restrict__ emb,
                  const float* __restrict__ fb, const float* __restrict__ scale,
                  const float* __restrict__ bias, float* __restrict__ out, int B)
{
  int row  = blockIdx.x * 8 + (threadIdx.x >> 5);
  int lane = threadIdx.x & 31;
  if (row >= B) return;
  int id = idx[row];
  const float* src = (id >= 0) ? (emb + (size_t)id * GNN) : fb;
  float v[8], s = 0.f, s2 = 0.f;
  #pragma unroll
  for (int i = 0; i < 8; ++i) {
    v[i] = src[lane + i * 32];
    s += v[i]; s2 += v[i] * v[i];
  }
  #pragma unroll
  for (int off = 16; off > 0; off >>= 1) {
    s  += __shfl_xor(s,  off, 32);
    s2 += __shfl_xor(s2, off, 32);
  }
  float mu  = s  * (1.f / GNN);
  float var = s2 * (1.f / GNN) - mu * mu;
  float rs  = rsqrtf(var + 1e-5f);
  #pragma unroll
  for (int i = 0; i < 8; ++i) {
    int c = lane + i * 32;
    out[(size_t)row * GNN + c] = (v[i] - mu) * rs * scale[c] + bias[c];
  }
}

// LayerNorm(512). One wave per row, 16 elems/lane.
__global__ __launch_bounds__(256)
void ln512(const float* __restrict__ x, const float* __restrict__ scale,
           const float* __restrict__ bias, float* __restrict__ out, int B)
{
  int row  = blockIdx.x * 8 + (threadIdx.x >> 5);
  int lane = threadIdx.x & 31;
  if (row >= B) return;
  const float* src = x + (size_t)row * HDIM;
  float v[16], s = 0.f, s2 = 0.f;
  #pragma unroll
  for (int i = 0; i < 16; ++i) {
    v[i] = src[lane + i * 32];
    s += v[i]; s2 += v[i] * v[i];
  }
  #pragma unroll
  for (int off = 16; off > 0; off >>= 1) {
    s  += __shfl_xor(s,  off, 32);
    s2 += __shfl_xor(s2, off, 32);
  }
  float mu  = s  * (1.f / HDIM);
  float var = s2 * (1.f / HDIM) - mu * mu;
  float rs  = rsqrtf(var + 1e-5f);
  #pragma unroll
  for (int i = 0; i < 16; ++i) {
    int c = lane + i * 32;
    out[(size_t)row * HDIM + c] = (v[i] - mu) * rs * scale[c] + bias[c];
  }
}

static inline int cdiv(long a, long b) { return (int)((a + b - 1) / b); }

extern "C" void kernel_launch(void* const* d_in, const int* in_sizes, int n_in,
                              void* d_out, int out_size, void* d_ws, size_t ws_size,
                              hipStream_t stream)
{
  const int*   node_idx = (const int*)  d_in[0];
  const int*   eidx     = (const int*)  d_in[1];
  const float* ew       = (const float*)d_in[2];
  const float* cache    = (const float*)d_in[3];
  const float* w_root   = (const float*)d_in[4];
  const float* w_neigh  = (const float*)d_in[5];
  const float* w_post   = (const float*)d_in[6];
  const float* b_post   = (const float*)d_in[7];
  const float* fb       = (const float*)d_in[8];
  const float* ln_in_s  = (const float*)d_in[9];
  const float* ln_in_b  = (const float*)d_in[10];
  const float* w_in     = (const float*)d_in[11];
  const float* rb_ln_s  = (const float*)d_in[12];
  const float* rb_ln_b  = (const float*)d_in[13];
  const float* rb_fc1   = (const float*)d_in[14];
  const float* rb_fc2   = (const float*)d_in[15];
  const float* w_bil    = (const float*)d_in[16];
  const float* gene_emb = (const float*)d_in[17];
  float* out = (float*)d_out;

  const int  B  = in_sizes[0];
  const long E  = in_sizes[1] / 2;
  const int  NN = in_sizes[3] / GNN;

  // Workspace layout (floats); total ~23M floats (~92 MB), buffers recycled.
  float* ws = (float*)d_ws;
  const size_t NNG = (size_t)NN * GNN;
  float* agg  = ws;                 // [NN,256]
  float* t1   = ws + NNG;           // [NN,256]  cache@w_root^T
  float* xbuf = ws + 2 * NNG;       // [NN,256]  relu-combined
  float* emb  = agg;                // node_emb reuses agg
  float* pert = t1;                 // [B,256]   reuses t1
  float* hA   = xbuf;               // [B,512]
  float* hB   = xbuf + 2200000;     // [B,512]   still inside node-buf region
  float* t0   = t1;                 // [B,512]   LN scratch
  float* mid  = ws + 3 * NNG;       // [B,2048]  fc1 output
  float* blin = mid;                // [B,1536]  reuses mid after last block

  dim3 blk(256);

  // 1) agg = 0 ; agg[dst] += cache[src]*w
  zero_f32<<<cdiv((long)NNG, 256), 256, 0, stream>>>(agg, (long)NNG);
  long total = E * GNN;
  scatter_add<<<cdiv(total, 256), 256, 0, stream>>>(eidx, eidx + E, ew, cache, agg, total);

  // 2) graph tail: t1 = cache@w_root^T ; x = relu(agg@w_neigh^T + t1) ; emb = x@w_post^T + b
  dim3 g1(cdiv(GNN, 64), cdiv(NN, 64));
  linear_wmma<<<g1, blk, 0, stream>>>(cache, GNN, w_root,  nullptr, nullptr, 0, t1,   GNN, NN, GNN, GNN, 0);
  linear_wmma<<<g1, blk, 0, stream>>>(agg,   GNN, w_neigh, nullptr, t1,    GNN, xbuf, GNN, NN, GNN, GNN, 1);
  linear_wmma<<<g1, blk, 0, stream>>>(xbuf,  GNN, w_post,  b_post,  nullptr, 0, emb,  GNN, NN, GNN, GNN, 0);

  // 3) gather(+fallback) + LN(256) ; input_proj 256->512
  gather_ln256<<<cdiv(B, 8), 256, 0, stream>>>(node_idx, emb, fb, ln_in_s, ln_in_b, pert, B);
  dim3 g2(cdiv(HDIM, 64), cdiv(B, 64));
  linear_wmma<<<g2, blk, 0, stream>>>(pert, GNN, w_in, nullptr, nullptr, 0, hA, HDIM, B, HDIM, GNN, 0);

  // 4) 6 residual blocks: h = gelu(LN(h)@fc1^T)@fc2^T + h   (ping-pong h)
  float* hc = hA; float* hn = hB;
  dim3 gm(cdiv(MIDD, 64), cdiv(B, 64));
  for (int i = 0; i < NLAYER; ++i) {
    ln512<<<cdiv(B, 8), 256, 0, stream>>>(hc, rb_ln_s + (size_t)i * HDIM, rb_ln_b + (size_t)i * HDIM, t0, B);
    linear_wmma<<<gm, blk, 0, stream>>>(t0,  HDIM, rb_fc1 + (size_t)i * MIDD * HDIM, nullptr, nullptr, 0,
                                        mid, MIDD, B, MIDD, HDIM, 2);
    linear_wmma<<<g2, blk, 0, stream>>>(mid, MIDD, rb_fc2 + (size_t)i * HDIM * MIDD, nullptr, hc, HDIM,
                                        hn,  HDIM, B, HDIM, MIDD, 0);
    float* tsw = hc; hc = hn; hn = tsw;
  }

  // 5) bilinear head: blin = h@w_bil^T [B,1536]; logits_c = blin_c @ gene_emb^T [B,6640]
  dim3 g3(cdiv(NCLS * RNK, 64), cdiv(B, 64));
  linear_wmma<<<g3, blk, 0, stream>>>(hc, HDIM, w_bil, nullptr, nullptr, 0, blin, NCLS * RNK, B, NCLS * RNK, HDIM, 0);
  dim3 g4(cdiv(NGENE, 64), cdiv(B, 64));
  for (int c = 0; c < NCLS; ++c) {
    linear_wmma<<<g4, blk, 0, stream>>>(blin + c * RNK, NCLS * RNK, gene_emb, nullptr, nullptr, 0,
                                        out + (size_t)c * NGENE, NCLS * NGENE, B, NGENE, RNK, 0);
  }
  (void)n_in; (void)out_size; (void)ws_size;
}
